// PyramidMahalanobisMask_4174708212240
// MI455X (gfx1250) — matrix operand order repaired and checked by reading, not compiled, not run on hardware
//
#include <hip/hip_runtime.h>
#include <hip/hip_bf16.h>

// Problem constants
static constexpr int kL    = 2048;   // signal length
static constexpr int kBC   = 2048;   // B*C rows
static constexpr int kB    = 32;
static constexpr int kC    = 64;
static constexpr int kNPad = 352;    // padded band size (22*16)
static constexpr int kTotK = 3 * kNPad; // 1056

typedef __attribute__((ext_vector_type(16))) __bf16 v16bf;
typedef __attribute__((ext_vector_type(8)))  float  v8f;

union FragU { unsigned int u[8]; uint4 q[2]; v16bf v; };

__device__ __forceinline__ void split_bf(float x, unsigned short& h, unsigned short& l) {
  unsigned int bx = __float_as_uint(x);
  unsigned short hs = (unsigned short)(bx >> 16);           // truncate to bf16
  float fh = __uint_as_float(((unsigned int)hs) << 16);
  float r = x - fh;                                         // residual captured in lo
  h = hs;
  l = (unsigned short)(__float_as_uint(r) >> 16);
}

// ---------------------------------------------------------------------------
// Kernel 1: 2048-pt radix-2 DIT FFT per (b,c) row in LDS; write |rfft| split
// into bf16 hi/lo, band-major padded layout [row][band*352 + s], pads zeroed.
// Outputs are packed as uint (2 halfs) per store.
// ---------------------------------------------------------------------------
__global__ void __launch_bounds__(256)
fft_mag_kernel(const float* __restrict__ X,
               unsigned short* __restrict__ bandHi,
               unsigned short* __restrict__ bandLo) {
  __shared__ float re[kL];
  __shared__ float im[kL];
  const int row = blockIdx.x;
  const int tid = threadIdx.x;
  const float* xr = X + (size_t)row * kL;

  for (int i = tid; i < kL; i += 256) {
    int r = __brev((unsigned)i) >> 21;   // 11-bit reversal
    re[r] = xr[i];
    im[r] = 0.0f;
  }
  __syncthreads();

  for (int s = 1; s <= 11; ++s) {
    const int half = 1 << (s - 1);
    const int len  = 1 << s;
    for (int t = tid; t < 1024; t += 256) {
      int grp = t >> (s - 1);
      int k   = t & (half - 1);
      int i0  = grp * len + k;
      int i1  = i0 + half;
      float ang = -6.283185307179586f * (float)k / (float)len;
      float sn, cs;
      sincosf(ang, &sn, &cs);
      float r1 = re[i1], m1 = im[i1];
      float tr = cs * r1 - sn * m1;
      float ti = cs * m1 + sn * r1;
      float r0 = re[i0], m0 = im[i0];
      re[i1] = r0 - tr; im[i1] = m0 - ti;
      re[i0] = r0 + tr; im[i0] = m0 + ti;
    }
    __syncthreads();
  }

  unsigned int* outH = (unsigned int*)bandHi;
  unsigned int* outL = (unsigned int*)bandLo;
  for (int jj = tid; jj < kTotK / 2; jj += 256) {   // pairs never straddle a band
    unsigned short hh[2], ll[2];
#pragma unroll
    for (int e = 0; e < 2; ++e) {
      int j    = jj * 2 + e;
      int band = j / kNPad;
      int s    = j - band * kNPad;
      int bs   = (band == 2) ? 343 : 341;
      float m = 0.0f;
      if (s < bs) {
        int k = band * 341 + s;        // rfft bin 0..1024
        m = sqrtf(re[k] * re[k] + im[k] * im[k]);
      }
      split_bf(m, hh[e], ll[e]);
    }
    outH[(size_t)row * (kTotK / 2) + jj] = (unsigned int)hh[0] | ((unsigned int)hh[1] << 16);
    outL[(size_t)row * (kTotK / 2) + jj] = (unsigned int)ll[0] | ((unsigned int)ll[1] << 16);
  }
}

// ---------------------------------------------------------------------------
// Kernel 2: Ap[band][d][s] = A_band[d][s], zero-padded to 352x352, bf16 hi/lo.
// (This row-major padded copy IS the transposed-staging layout the WMMA
//  B-fragment wants: Bt[s][d]^T = A[d][s].)
// ---------------------------------------------------------------------------
__global__ void __launch_bounds__(256)
prep_a_kernel(const float* __restrict__ A0, const float* __restrict__ A1,
              const float* __restrict__ A2,
              unsigned short* __restrict__ ApHi, unsigned short* __restrict__ ApLo) {
  int idx = blockIdx.x * 256 + threadIdx.x;
  const int total = 3 * kNPad * kNPad;
  if (idx >= total) return;
  int band = idx / (kNPad * kNPad);
  int rem  = idx - band * (kNPad * kNPad);
  int d    = rem / kNPad;
  int s    = rem - d * kNPad;
  int bs   = (band == 2) ? 343 : 341;
  const float* A = (band == 0) ? A0 : (band == 1) ? A1 : A2;
  float v = (d < bs && s < bs) ? A[d * bs + s] : 0.0f;
  unsigned short h, l;
  split_bf(v, h, l);
  ApHi[idx] = h; ApLo[idx] = l;
}

// ---------------------------------------------------------------------------
// Kernel 3: Y[row,d] = sum_s band[row,s] * A[d,s], scaled by sqrt(softmax(w)).
// M=2048 rows, N=K=352 per band. bf16 2-split -> 3 WMMAs per tile per K-step.
// Block: 128 threads (4 waves), tile M=64, N=176 (11 N-tiles per wave).
// All fragments load as 2x ds_load_b128 per operand.
// ---------------------------------------------------------------------------
__global__ void __launch_bounds__(128)
gemm_y_kernel(const unsigned short* __restrict__ bandHi,
              const unsigned short* __restrict__ bandLo,
              const unsigned short* __restrict__ ApHi,
              const unsigned short* __restrict__ ApLo,
              const float* __restrict__ fusion_w,
              unsigned short* __restrict__ Yhi,
              unsigned short* __restrict__ Ylo) {
  const int mblk = blockIdx.x;   // 0..31
  const int nblk = blockIdx.y;   // 0..1
  const int band = blockIdx.z;   // 0..2
  const int tid  = threadIdx.x;
  const int wave = tid >> 5;
  const int lane = tid & 31;
  const int h    = lane >> 4;    // half-wave select
  const int ml   = lane & 15;

  __shared__ __align__(16) unsigned short aH[64 * 32],  aL[64 * 32];   // [m][k]
  __shared__ __align__(16) unsigned short bH[176 * 32], bL[176 * 32];  // [n][k] (B transposed)

  // softmax over 3 fusion weights; fold sqrt(w_band) into Y.
  float f0 = fusion_w[0], f1 = fusion_w[1], f2 = fusion_w[2];
  float fm = fmaxf(f0, fmaxf(f1, f2));
  float e0 = __expf(f0 - fm), e1 = __expf(f1 - fm), e2 = __expf(f2 - fm);
  float wsel = (band == 0) ? e0 : (band == 1) ? e1 : e2;
  float sw = sqrtf(wsel / (e0 + e1 + e2));

  const v8f vz = {0,0,0,0,0,0,0,0};
  v8f acc[11];
#pragma unroll
  for (int t = 0; t < 11; ++t) acc[t] = vz;

  const int n0 = nblk * 176;
  const int rowbase = mblk * 64;
  const unsigned short* apH = ApHi + (size_t)band * kNPad * kNPad;
  const unsigned short* apL = ApLo + (size_t)band * kNPad * kNPad;

  for (int kk = 0; kk < kNPad; kk += 32) {
    __syncthreads();
    // Stage A slab [64][32] (vectorized: 8 halfs per chunk, 256 chunks)
    for (int c4 = tid; c4 < 64 * 4; c4 += 128) {
      int r = c4 >> 2, c8 = (c4 & 3) << 3;
      size_t g = (size_t)(rowbase + r) * kTotK + band * kNPad + kk + c8;
      *(uint4*)(aH + r * 32 + c8) = *(const uint4*)(bandHi + g);
      *(uint4*)(aL + r * 32 + c8) = *(const uint4*)(bandLo + g);
    }
    // Stage B slab transposed [176][32]: bT[n][k] = A[n0+n][kk+k] (contiguous rows)
    for (int c4 = tid; c4 < 176 * 4; c4 += 128) {
      int n = c4 >> 2, k8 = (c4 & 3) << 3;
      size_t g = (size_t)(n0 + n) * kNPad + kk + k8;
      *(uint4*)(bH + n * 32 + k8) = *(const uint4*)(apH + g);
      *(uint4*)(bL + n * 32 + k8) = *(const uint4*)(apL + g);
    }
    // Prefetch next K-slab while this one computes
    if (kk + 32 < kNPad) {
      size_t ga = (size_t)(rowbase + (tid & 63)) * kTotK + band * kNPad + kk + 32;
      __builtin_prefetch(bandHi + ga, 0, 1);
      __builtin_prefetch(bandLo + ga, 0, 1);
      size_t gb = (size_t)(n0 + (tid >= 64 ? tid + 48 : tid)) * kNPad + kk + 32;
      __builtin_prefetch(apH + gb, 0, 1);
      __builtin_prefetch(apL + gb, 0, 1);
    }
    __syncthreads();

    // A fragment (16x32): two contiguous 16B chunks per lane
    FragU fah, fal;
    const int mofs = (wave * 16 + ml) * 32 + (h << 3);
    fah.q[0] = *(const uint4*)(aH + mofs);
    fah.q[1] = *(const uint4*)(aH + mofs + 16);
    fal.q[0] = *(const uint4*)(aL + mofs);
    fal.q[1] = *(const uint4*)(aL + mofs + 16);

#pragma unroll
    for (int t = 0; t < 11; ++t) {
      // B fragment (32x16): contiguous in the [n][k] slab
      FragU fbh, fbl;
      const int nofs = (t * 16 + ml) * 32 + (h << 3);
      fbh.q[0] = *(const uint4*)(bH + nofs);
      fbh.q[1] = *(const uint4*)(bH + nofs + 16);
      fbl.q[0] = *(const uint4*)(bL + nofs);
      fbl.q[1] = *(const uint4*)(bL + nofs + 16);

      acc[t] = __builtin_amdgcn_wmma_f32_16x16x32_bf16(false, fah.v, false, fbh.v, (short)0, acc[t], false, false);
      acc[t] = __builtin_amdgcn_wmma_f32_16x16x32_bf16(false, fah.v, false, fbl.v, (short)0, acc[t], false, false);
      acc[t] = __builtin_amdgcn_wmma_f32_16x16x32_bf16(false, fal.v, false, fbh.v, (short)0, acc[t], false, false);
    }
  }

  // Store Y scaled by sqrt(w_band), re-split to bf16 hi/lo for the Gram GEMM.
#pragma unroll
  for (int t = 0; t < 11; ++t) {
#pragma unroll
    for (int r = 0; r < 8; ++r) {
      int row = rowbase + wave * 16 + h * 8 + r;
      int col = band * kNPad + n0 + t * 16 + ml;
      float v = acc[t][r] * sw;
      unsigned short hh, ll;
      split_bf(v, hh, ll);
      Yhi[(size_t)row * kTotK + col] = hh;
      Ylo[(size_t)row * kTotK + col] = ll;
    }
  }
}

// ---------------------------------------------------------------------------
// Kernel 4: per batch b: G = Y_b Y_b^T (64x64, K=1056) fused with the full
// epilogue: dist -> 1/(dist+eps) -> row max -> p -> logits -> gumbel argmax.
// 256 threads (8 waves), 16 output tiles of 16x16 -> 2 tiles per wave.
// A and B fragments both come from the same row-major Y slab (b128 loads).
// ---------------------------------------------------------------------------
__global__ void __launch_bounds__(256)
gram_sample_kernel(const unsigned short* __restrict__ Yhi,
                   const unsigned short* __restrict__ Ylo,
                   const float* __restrict__ gumbel,
                   float* __restrict__ out) {
  const int b    = blockIdx.x;
  const int tid  = threadIdx.x;
  const int wave = tid >> 5;
  const int lane = tid & 31;
  const int h    = lane >> 4;
  const int nl   = lane & 15;

  __shared__ __align__(16) unsigned short yH[64 * 32], yL[64 * 32];
  __shared__ float gmat[64 * 64];
  __shared__ float pm[64 * 4];

  const v8f vz = {0,0,0,0,0,0,0,0};
  v8f acc[2] = {vz, vz};
  const int mis[2] = { (wave * 2 + 0) >> 2, (wave * 2 + 1) >> 2 };
  const int nis[2] = { (wave * 2 + 0) & 3,  (wave * 2 + 1) & 3  };

  for (int kk = 0; kk < kTotK; kk += 32) {
    __syncthreads();
    // Stage Y slab [64][32]: exactly one uint4 pair per thread
    {
      int r = tid >> 2, c8 = (tid & 3) << 3;
      size_t g = (size_t)(b * 64 + r) * kTotK + kk + c8;
      *(uint4*)(yH + r * 32 + c8) = *(const uint4*)(Yhi + g);
      *(uint4*)(yL + r * 32 + c8) = *(const uint4*)(Ylo + g);
      if (kk + 32 < kTotK) {
        __builtin_prefetch(Yhi + g + 32, 0, 1);
        __builtin_prefetch(Ylo + g + 32, 0, 1);
      }
    }
    __syncthreads();

#pragma unroll
    for (int q = 0; q < 2; ++q) {
      FragU fah, fal, fbh, fbl;
      const int mofs = (mis[q] * 16 + nl) * 32 + (h << 3);   // A frag: row m
      const int nofs = (nis[q] * 16 + nl) * 32 + (h << 3);   // B frag: B[k][n]=Y[n][k]
      fah.q[0] = *(const uint4*)(yH + mofs);
      fah.q[1] = *(const uint4*)(yH + mofs + 16);
      fal.q[0] = *(const uint4*)(yL + mofs);
      fal.q[1] = *(const uint4*)(yL + mofs + 16);
      fbh.q[0] = *(const uint4*)(yH + nofs);
      fbh.q[1] = *(const uint4*)(yH + nofs + 16);
      fbl.q[0] = *(const uint4*)(yL + nofs);
      fbl.q[1] = *(const uint4*)(yL + nofs + 16);

      acc[q] = __builtin_amdgcn_wmma_f32_16x16x32_bf16(false, fah.v, false, fbh.v, (short)0, acc[q], false, false);
      acc[q] = __builtin_amdgcn_wmma_f32_16x16x32_bf16(false, fah.v, false, fbl.v, (short)0, acc[q], false, false);
      acc[q] = __builtin_amdgcn_wmma_f32_16x16x32_bf16(false, fal.v, false, fbh.v, (short)0, acc[q], false, false);
    }
  }
  __syncthreads();

#pragma unroll
  for (int q = 0; q < 2; ++q) {
#pragma unroll
    for (int r = 0; r < 8; ++r) {
      gmat[(mis[q] * 16 + h * 8 + r) * 64 + nis[q] * 16 + nl] = acc[q][r];
    }
  }
  __syncthreads();

  // Row-max of exp_dist (4 partials per row).
  {
    int row = tid >> 2, part = tid & 3;
    float qi = gmat[row * 64 + row];
    float mx = 0.0f;
    for (int j = part; j < 64; j += 4) {
      if (j == row) continue;
      float qj = gmat[j * 64 + j];
      float dist = fmaxf(qi + qj - 2.0f * gmat[row * 64 + j], 0.0f);
      mx = fmaxf(mx, 1.0f / (dist + 1e-10f));
    }
    pm[row * 4 + part] = mx;
  }
  __syncthreads();

  // Final: p -> logits -> gumbel argmax -> hard sample (forward value of ST).
  for (int idx = tid; idx < 64 * 64; idx += 256) {
    int i = idx >> 6, j = idx & 63;
    float p;
    if (i == j) {
      p = 0.99f;
    } else {
      float rmax = fmaxf(fmaxf(pm[i * 4 + 0], pm[i * 4 + 1]),
                         fmaxf(pm[i * 4 + 2], pm[i * 4 + 3]));
      float qi = gmat[i * 64 + i], qj = gmat[j * 64 + j];
      float dist = fmaxf(qi + qj - 2.0f * gmat[i * 64 + j], 0.0f);
      float ed = 1.0f / (dist + 1e-10f);
      p = (ed / rmax) * 0.99f;
    }
    float l0 = logf(p / (1.0f - p));
    const float* g2 = gumbel + (((size_t)(b * 64 + i) * 64 + j) * 2);
    float y0 =  l0 + g2[0];
    float y1 = -l0 + g2[1];
    out[(size_t)b * 4096 + idx] = (y0 >= y1) ? 1.0f : 0.0f;  // argmax==0 -> 1.0
  }
}

// ---------------------------------------------------------------------------
extern "C" void kernel_launch(void* const* d_in, const int* in_sizes, int n_in,
                              void* d_out, int out_size, void* d_ws, size_t ws_size,
                              hipStream_t stream) {
  const float* X  = (const float*)d_in[0];
  const float* A0 = (const float*)d_in[1];
  const float* A1 = (const float*)d_in[2];
  const float* A2 = (const float*)d_in[3];
  const float* fw = (const float*)d_in[4];
  const float* gn = (const float*)d_in[5];
  float* out = (float*)d_out;

  char* ws = (char*)d_ws;
  const size_t bandBytes = (size_t)kBC * kTotK * sizeof(unsigned short);   // 4.3 MB
  const size_t apBytes   = (size_t)3 * kNPad * kNPad * sizeof(unsigned short);

  unsigned short* bandHi = (unsigned short*)ws;              ws += bandBytes;
  unsigned short* bandLo = (unsigned short*)ws;              ws += bandBytes;
  unsigned short* ApHi   = (unsigned short*)ws;              ws += apBytes;
  unsigned short* ApLo   = (unsigned short*)ws;              ws += apBytes;
  unsigned short* Yhi    = (unsigned short*)ws;              ws += bandBytes;
  unsigned short* Ylo    = (unsigned short*)ws;              ws += bandBytes;

  fft_mag_kernel<<<kBC, 256, 0, stream>>>(X, bandHi, bandLo);

  const int totalAp = 3 * kNPad * kNPad;
  prep_a_kernel<<<(totalAp + 255) / 256, 256, 0, stream>>>(A0, A1, A2, ApHi, ApLo);

  gemm_y_kernel<<<dim3(32, 2, 3), 128, 0, stream>>>(bandHi, bandLo, ApHi, ApLo,
                                                    fw, Yhi, Ylo);

  gram_sample_kernel<<<kB, 256, 0, stream>>>(Yhi, Ylo, gn, out);
}
